// RationaleConceptBottleneckModel_40810779247515
// MI455X (gfx1250) — matrix operand order repaired and checked by reading, not compiled, not run on hardware
//
#include <hip/hip_runtime.h>
#include <hip/hip_bf16.h>
#include <math.h>

// ---------------------------------------------------------------------------
// RationaleConceptBottleneckModel for MI455X (gfx1250), bf16 WMMA pipeline.
// B=16, S=2048, H=768, HH=384, C=64, L=4
// Fragments: 2x ds_load_b128 per operand (A row-major, B transposed staging).
// Staging: async global->LDS (ASYNCcnt) with double buffering, last iter peeled.
// ---------------------------------------------------------------------------

typedef __attribute__((ext_vector_type(16))) __bf16 v16bf;
typedef __attribute__((ext_vector_type(8)))  float  v8f;
typedef __attribute__((ext_vector_type(4)))  int    v4i;

#define B_   16
#define S_   2048
#define H_   768
#define HH_  384
#define C_   64
#define L_   4

// padded leading dims (elements); byte strides stay multiples of 16
#define QS_LD   392   // 784 B rows, bank step 4
#define KS_LD   392
#define P_LD    2056  // 4112 B rows, bank step 4
#define XT_LD   40    // 80 B rows, bank step 20
#define GA_LD   40
#define GB_LD   40

// Attention LDS:
//   qs     : 16  x QS_LD bf16 = 12544 B
//   scores : 16  x 2048  f32  = 131072 B   (reused as 2x x-tile in phase 3)
//   stage  : max(k-tile 128 x KS_LD bf16 = 100352,
//                p 16 x P_LD bf16 = 65792) -> region sized 127232
#define ATTN_Q_BYTES   12544
#define ATTN_SC_BYTES  131072
#define ATTN_P_BYTES   65792
#define XSTAGE_BYTES   (768 * XT_LD * 2)  // 61440, two of them fit in scores
#define ATTN_LDS_BYTES (ATTN_Q_BYTES + ATTN_SC_BYTES + 127232)  // 270848

// ----------------------- async global->LDS helpers -------------------------

#if __has_builtin(__builtin_amdgcn_global_load_async_to_lds_b128) && \
    __has_builtin(__builtin_amdgcn_s_wait_asynccnt)
#define ASYNC_LDS 1
typedef __attribute__((address_space(1))) v4i gv4i_t;  // global v4i
typedef __attribute__((address_space(3))) v4i lv4i_t;  // LDS v4i
#endif

// copy 16 aligned bytes global -> LDS
__device__ __forceinline__ void cp16(void* lds, const void* g) {
#ifdef ASYNC_LDS
  __builtin_amdgcn_global_load_async_to_lds_b128((gv4i_t*)g, (lv4i_t*)lds, 0, 0);
#else
  *(uint4*)lds = *(const uint4*)g;
#endif
}

template <int N>
__device__ __forceinline__ void cp_wait() {
#ifdef ASYNC_LDS
  __builtin_amdgcn_s_wait_asynccnt(N);
#endif
}

// ------------------------------- helpers -----------------------------------

__device__ __forceinline__ float wred_sum(float v) {
#pragma unroll
  for (int o = 16; o > 0; o >>= 1) v += __shfl_xor(v, o, 32);
  return v;
}
__device__ __forceinline__ float wred_max(float v) {
#pragma unroll
  for (int o = 16; o > 0; o >>= 1) v = fmaxf(v, __shfl_xor(v, o, 32));
  return v;
}

// A fragment: 16x32 bf16 (MxK) from row-major LDS (row stride %16B==0).
__device__ __forceinline__ v16bf load_a_frag(const __bf16* a, int ld, int row0,
                                             int k0, int lane) {
  const int m  = lane & 15;
  const int hk = (lane >> 4) << 3;  // 0 or 8
  const __bf16* p = a + (size_t)(row0 + m) * ld + k0 + hk;
  union { uint4 u[2]; v16bf v; } t;
  t.u[0] = *(const uint4*)(p);
  t.u[1] = *(const uint4*)(p + 16);
  return t.v;
}

// B fragment from TRANSPOSED staging Bt[n][k] (row stride %16B==0).
__device__ __forceinline__ v16bf load_bt_frag(const __bf16* bt, int ld, int k0,
                                              int n0, int lane) {
  const __bf16* p = bt + (size_t)(n0 + (lane & 15)) * ld + k0 + ((lane >> 4) << 3);
  union { uint4 u[2]; v16bf v; } t;
  t.u[0] = *(const uint4*)(p);
  t.u[1] = *(const uint4*)(p + 16);
  return t.v;
}

// --------------------------- f32 -> bf16 cast ------------------------------

__global__ void cvt_bf16_kernel(const float* __restrict__ src,
                                __bf16* __restrict__ dst, long n) {
  long i = (long)blockIdx.x * blockDim.x + threadIdx.x;
  long stride = (long)gridDim.x * blockDim.x;
  for (; i < n; i += stride) dst[i] = (__bf16)src[i];
}

// ----------------- f32 -> bf16 transpose: dst[b][c][r] = src[b][r][c] ------

__global__ __launch_bounds__(256) void cvt_tr_kernel(
    const float* __restrict__ src, __bf16* __restrict__ dst, int R, int Cc) {
  __shared__ float tile[32][33];
  const int tx = threadIdx.x & 31, ty = threadIdx.x >> 5;
  const int c0 = blockIdx.x * 32, r0 = blockIdx.y * 32, b = blockIdx.z;
#pragma unroll
  for (int i = 0; i < 4; ++i)
    tile[ty + 8 * i][tx] =
        src[((size_t)b * R + r0 + ty + 8 * i) * Cc + c0 + tx];
  __syncthreads();
#pragma unroll
  for (int i = 0; i < 4; ++i)
    dst[((size_t)b * Cc + c0 + ty + 8 * i) * R + r0 + tx] =
        (__bf16)tile[tx][ty + 8 * i];
}

// ------------------------------ WMMA GEMM ----------------------------------
// C[M,N] = A[M,K](bf16 rm) @ Bt[N,K](bf16 pre-transposed) + bias[N]
// mode 0: f32 out; mode 1: bf16 out. Tile 128x64/WG, 8 waves, double-buffered
// async staging with the final k-step peeled (branch-free steady state).

__global__ __launch_bounds__(256) void gemm_bf16_kernel(
    const __bf16* __restrict__ A, const __bf16* __restrict__ Bt,
    const float* __restrict__ bias, void* __restrict__ out,
    int M, int N, int K, int mode) {
  __shared__ __bf16 As[2][128 * GA_LD];
  __shared__ __bf16 Bs[2][64 * GB_LD];
  const int tid  = threadIdx.x;
  const int lane = tid & 31;
  const int wave = tid >> 5;
  const int row0g = blockIdx.x * 128;
  const int n0g   = blockIdx.y * 64;

  v8f acc[4];
#pragma unroll
  for (int j = 0; j < 4; ++j)
#pragma unroll
    for (int r = 0; r < 8; ++r) acc[j][r] = 0.0f;

  const int arow = tid >> 1, ahalf = tid & 1;  // 128 rows x 2 16B halves
  const int brow = tid >> 2, bch   = tid & 3;  // 64 rows x 4 16B chunks
  const __bf16* agp = A  + (size_t)(row0g + arow) * K;
  const __bf16* bgp = Bt + (size_t)(n0g + brow) * K;

  auto issue = [&](int buf, int k0) {
    cp16(&As[buf][arow * GA_LD + ahalf * 8], agp + k0 + ahalf * 8);
    cp16(&Bs[buf][brow * GB_LD + bch * 8],  bgp + k0 + bch * 8);
  };
  auto compute = [&](int buf) {
    const __bf16* Ab = As[buf];
    const __bf16* Bb = Bs[buf];
    v16bf af = load_a_frag(Ab, GA_LD, wave * 16, 0, lane);
#pragma unroll
    for (int j = 0; j < 4; ++j) {
      v16bf bf = load_bt_frag(Bb, GB_LD, 0, j * 16, lane);
      acc[j] = __builtin_amdgcn_wmma_f32_16x16x32_bf16(
          false, af, false, bf, (short)0, acc[j], false, false);
    }
  };

  const int nk = K >> 5;
  issue(0, 0);
  for (int i = 0; i < nk - 1; ++i) {   // steady state: branch-free
    issue((i + 1) & 1, (i + 1) << 5);
    cp_wait<2>();
    __syncthreads();  // all waves' stage-i data in LDS
    compute(i & 1);
    __syncthreads();  // stage-i buffer free for reuse at i+2
  }
  cp_wait<0>();       // peeled last k-step
  __syncthreads();
  compute((nk - 1) & 1);

  const int nlo = lane & 15, mhi = (lane >> 4) * 8;
#pragma unroll
  for (int j = 0; j < 4; ++j) {
    const int n  = n0g + j * 16 + nlo;
    const float bv = bias[n];
#pragma unroll
    for (int r = 0; r < 8; ++r) {
      const int m = row0g + wave * 16 + mhi + r;
      const float v = acc[j][r] + bv;
      if (mode == 0) ((float*)out)[(size_t)m * N + n] = v;
      else           ((__bf16*)out)[(size_t)m * N + n] = (__bf16)v;
    }
  }
}

// --------------------------- fused attention -------------------------------
// One WG per (batch, 16-row block). Scores fully LDS-resident (16x2048 f32).

__global__ __launch_bounds__(256) void attn_kernel(
    const __bf16* __restrict__ qbf, const __bf16* __restrict__ kbf,
    const __bf16* __restrict__ xtbf, const int* __restrict__ am,
    __bf16* __restrict__ ctxbf) {
  extern __shared__ char smem[];
  __bf16* qs    = (__bf16*)smem;                                   // 16 x QS_LD
  float*  sc    = (float*)(smem + ATTN_Q_BYTES);                   // 16 x 2048
  __bf16* stage = (__bf16*)(smem + ATTN_Q_BYTES + ATTN_SC_BYTES);  // k-tile / p

  const int tid = threadIdx.x, lane = tid & 31, wave = tid >> 5;
  const int b  = blockIdx.x >> 7;
  const int s0 = (blockIdx.x & 127) << 4;

  // stage q block: 16 rows x 48 chunks of 16B (async; covered by first wait)
  for (int i = tid; i < 16 * 48; i += 256) {
    const int row = i / 48, ch = i % 48;
    cp16((char*)qs + row * (QS_LD * 2) + ch * 16,
         (const char*)(qbf + (size_t)(b * S_ + s0 + row) * HH_) + ch * 16);
  }

  const float scale = 0.05103103630798288f;  // 1/sqrt(384)

  // Phase 1: scores = scale * q @ k^T  (Bt of this GEMM == raw k rows)
  for (int tc = 0; tc < S_; tc += 128) {
    __syncthreads();  // previous chunk's compute done; stage reusable
    for (int i = tid; i < 128 * 48; i += 256) {
      const int row = i / 48, ch = i % 48;
      cp16((char*)stage + row * (KS_LD * 2) + ch * 16,
           (const char*)(kbf + (size_t)(b * S_ + tc + row) * HH_) + ch * 16);
    }
    cp_wait<0>();
    __syncthreads();
    v8f acc;
#pragma unroll
    for (int r = 0; r < 8; ++r) acc[r] = 0.0f;
#pragma unroll
    for (int ks = 0; ks < 12; ++ks) {
      v16bf af = load_a_frag(qs, QS_LD, 0, ks * 32, lane);
      v16bf bf = load_bt_frag(stage, KS_LD, ks * 32, wave * 16, lane);
      acc = __builtin_amdgcn_wmma_f32_16x16x32_bf16(
          false, af, false, bf, (short)0, acc, false, false);
    }
    const int n = lane & 15, mh = (lane >> 4) * 8;
    const int t = tc + wave * 16 + n;
    const int amv = am[b * S_ + t];
#pragma unroll
    for (int r = 0; r < 8; ++r) {
      float v = acc[r] * scale;
      if (!amv) v = -10000.0f;
      sc[(mh + r) * S_ + t] = v;
    }
  }
  __syncthreads();

  // Phase 2: softmax (wave handles rows 2w, 2w+1); p -> stage (bf16, P_LD rows)
  for (int rr = 0; rr < 2; ++rr) {
    const int m = wave * 2 + rr;
    float mx = -3.0e38f;
    for (int j = 0; j < 64; ++j) mx = fmaxf(mx, sc[m * S_ + lane + 32 * j]);
    mx = wred_max(mx);
    float sm = 0.0f;
    for (int j = 0; j < 64; ++j) sm += __expf(sc[m * S_ + lane + 32 * j] - mx);
    sm = wred_sum(sm);
    const float inv = 1.0f / sm;
    for (int j = 0; j < 64; ++j) {
      const int t = lane + 32 * j;
      stage[m * P_LD + t] = (__bf16)(__expf(sc[m * S_ + t] - mx) * inv);
    }
  }
  __syncthreads();  // scores region now dead -> reuse as double-buffered x-tiles

  // Phase 3: ctx[16][768] = p @ x; x-tiles from xT[b][h][s], double-buffered
  __bf16* xb[2] = { (__bf16*)sc, (__bf16*)((char*)sc + XSTAGE_BYTES) };
  const __bf16* xg = xtbf + (size_t)b * H_ * S_;

  auto issue_x = [&](__bf16* dstbuf, int k0) {
    for (int i = tid; i < 768 * 4; i += 256) {  // 12 cp16 per thread
      const int row = i >> 2, ch = i & 3;
      cp16((char*)dstbuf + row * (XT_LD * 2) + ch * 16,
           (const char*)(xg + (size_t)row * S_ + k0) + ch * 16);
    }
  };

  v8f cacc[6];
#pragma unroll
  for (int j = 0; j < 6; ++j)
#pragma unroll
    for (int r = 0; r < 8; ++r) cacc[j][r] = 0.0f;

  auto compute_x = [&](const __bf16* xcur, int k0) {
    v16bf af = load_a_frag(stage, P_LD, 0, k0, lane);
#pragma unroll
    for (int j = 0; j < 6; ++j) {
      v16bf bf = load_bt_frag(xcur, XT_LD, 0, wave * 96 + j * 16, lane);
      cacc[j] = __builtin_amdgcn_wmma_f32_16x16x32_bf16(
          false, af, false, bf, (short)0, cacc[j], false, false);
    }
  };

  issue_x(xb[0], 0);
  for (int i = 0; i < 63; ++i) {   // steady state: branch-free
    issue_x(xb[(i + 1) & 1], (i + 1) << 5);
    cp_wait<12>();
    __syncthreads();
    compute_x(xb[i & 1], i << 5);
    __syncthreads();
  }
  cp_wait<0>();                    // peeled last k-step
  __syncthreads();
  compute_x(xb[63 & 1], 63 << 5);

  const int n15 = lane & 15, mh = (lane >> 4) * 8;
#pragma unroll
  for (int j = 0; j < 6; ++j) {
    const int n = wave * 96 + j * 16 + n15;
#pragma unroll
    for (int r = 0; r < 8; ++r)
      ctxbf[(size_t)(b * S_ + s0 + mh + r) * H_ + n] = (__bf16)cacc[j][r];
  }
}

// ------------------- LN + ReLU + W2 dot -> token scores --------------------

__global__ __launch_bounds__(256) void lnscore_kernel(
    const float* __restrict__ hpre, const float* __restrict__ g1,
    const float* __restrict__ be1, const float* __restrict__ W2,
    const float* __restrict__ b2, const int* __restrict__ am,
    float* __restrict__ ts) {
  const int lane = threadIdx.x & 31, wave = threadIdx.x >> 5;
  const int row = blockIdx.x * 8 + wave;
  const float* hp = hpre + (size_t)row * HH_;
  float v[12];
  float s = 0.0f;
#pragma unroll
  for (int j = 0; j < 12; ++j) { v[j] = hp[lane + 32 * j]; s += v[j]; }
  const float mean = wred_sum(s) * (1.0f / HH_);
  float s2 = 0.0f;
#pragma unroll
  for (int j = 0; j < 12; ++j) { const float d = v[j] - mean; s2 += d * d; }
  const float var = wred_sum(s2) * (1.0f / HH_);
  const float rs = rsqrtf(var + 1e-5f);
  float dot = 0.0f;
#pragma unroll
  for (int j = 0; j < 12; ++j) {
    const int i = lane + 32 * j;
    float h = (v[j] - mean) * rs * g1[i] + be1[i];
    h = fmaxf(h, 0.0f);
    dot += h * W2[i];
  }
  dot = wred_sum(dot);
  if (lane == 0) {
    float t = dot + b2[0];
    if (!am[row]) t = -10000.0f;
    ts[row] = t;
  }
}

// --------------------------- span mask search ------------------------------

__global__ __launch_bounds__(256) void span_kernel(
    const float* __restrict__ ts, const int* __restrict__ am,
    float* __restrict__ rmask) {
  __shared__ float cum[S_ + 1];
  __shared__ float bvs[256];
  __shared__ int   bis[256];
  __shared__ int   ired[256];
  __shared__ int   vlen_s;
  __shared__ int   besti_s;
  const int tid = threadIdx.x, b = blockIdx.x;

  int p = 0;
  for (int k = 0; k < 8; ++k) p += am[b * S_ + tid + 256 * k];
  ired[tid] = p;
  __syncthreads();
  if (tid == 0) {
    int t = 0;
    for (int i = 0; i < 256; ++i) t += ired[i];
    vlen_s = t;
    float c = 0.0f;
    cum[0] = 0.0f;
    for (int i = 0; i < S_; ++i) { c += ts[b * S_ + i]; cum[i + 1] = c; }
  }
  __syncthreads();
  const int vlen = vlen_s;

  float bestv = -3.0e38f;
  int besti = 0;
  for (int f = tid; f < 18 * S_; f += 256) {  // f = l_idx*S + start (span-major)
    const int li = f >> 11, st = f & (S_ - 1);
    const int Lsp = 3 + li;
    const int end = st + Lsp;
    const int endc = end > S_ ? S_ : end;
    float avg = (cum[endc] - cum[st]) / (float)Lsp + 0.01f * (float)Lsp / 20.0f;
    if (end > vlen) avg = -1.0e30f;
    if (avg > bestv) { bestv = avg; besti = f; }
  }
  bvs[tid] = bestv;
  bis[tid] = besti;
  __syncthreads();
  if (tid == 0) {
    float best = bvs[0];
    int idx = bis[0];
    for (int i = 1; i < 256; ++i)
      if (bvs[i] > best || (bvs[i] == best && bis[i] < idx)) {
        best = bvs[i];
        idx = bis[i];
      }
    besti_s = idx;
  }
  __syncthreads();
  const int idx = besti_s;
  const int Lsp = 3 + (idx >> 11);
  const int st = idx & (S_ - 1);
  const bool useShort = (vlen <= 3);
  for (int s = tid; s < S_; s += 256) {
    const bool m = useShort ? (s < vlen) : (s >= st && s < st + Lsp);
    rmask[b * S_ + s] = m ? 1.0f : 0.0f;
  }
}

// ------------------------------- pooling -----------------------------------

__global__ __launch_bounds__(256) void pool_kernel(
    const float* __restrict__ x, const float* __restrict__ rmask,
    float* __restrict__ pooled) {
  __shared__ float part[256];
  __shared__ float tot;
  const int tid = threadIdx.x, b = blockIdx.x;
  float p = 0.0f;
  for (int k = 0; k < 8; ++k) p += rmask[b * S_ + tid + 256 * k];
  part[tid] = p;
  __syncthreads();
  if (tid == 0) {
    float t = 0.0f;
    for (int i = 0; i < 256; ++i) t += part[i];
    tot = t;
  }
  __syncthreads();
  const float inv = 1.0f / (tot + 1e-6f);
  for (int c = tid; c < H_; c += 256) {
    float a = 0.0f;
    for (int s = 0; s < S_; ++s)
      a += x[((size_t)b * S_ + s) * H_ + c] * rmask[b * S_ + s];
    pooled[b * H_ + c] = a * inv;
  }
}

// ---------------------------- concept head ---------------------------------

__global__ __launch_bounds__(256) void head_kernel(
    const float* __restrict__ pooled, const float* __restrict__ x,
    const float* __restrict__ Wc1, const float* __restrict__ bc1,
    const float* __restrict__ g2, const float* __restrict__ be2,
    const float* __restrict__ Wc2, const float* __restrict__ bc2,
    const float* __restrict__ Mint, const float* __restrict__ Wcls,
    const float* __restrict__ bcls, float* __restrict__ out) {
  __shared__ float t1[16 * HH_];
  __shared__ float csb[16 * C_];
  __shared__ float symb[C_ * C_];
  const int tid = threadIdx.x, lane = tid & 31, wave = tid >> 5;

  for (int o = tid; o < 16 * HH_; o += 256) {
    const int bb = o / HH_, c = o % HH_;
    float a = bc1[c];
    for (int h = 0; h < H_; ++h) a += pooled[bb * H_ + h] * Wc1[h * HH_ + c];
    t1[o] = a;
  }
  __syncthreads();

  for (int rr = 0; rr < 2; ++rr) {
    const int row = wave * 2 + rr;
    float v[12];
    float s = 0.0f;
#pragma unroll
    for (int j = 0; j < 12; ++j) { v[j] = t1[row * HH_ + lane + 32 * j]; s += v[j]; }
    const float mean = wred_sum(s) * (1.0f / HH_);
    float s2 = 0.0f;
#pragma unroll
    for (int j = 0; j < 12; ++j) { const float d = v[j] - mean; s2 += d * d; }
    const float var = wred_sum(s2) * (1.0f / HH_);
    const float rsd = rsqrtf(var + 1e-5f);
#pragma unroll
    for (int j = 0; j < 12; ++j) {
      const int i = lane + 32 * j;
      t1[row * HH_ + i] = fmaxf((v[j] - mean) * rsd * g2[i] + be2[i], 0.0f);
    }
  }
  __syncthreads();

  for (int o = tid; o < 16 * C_; o += 256) {
    const int bb = o >> 6, c = o & 63;
    float a = bc2[c];
    for (int h = 0; h < HH_; ++h) a += t1[bb * HH_ + h] * Wc2[h * C_ + c];
    csb[o] = a;
  }
  for (int o = tid; o < C_ * C_; o += 256) {
    const int i = o >> 6, j = o & 63;
    symb[o] = 1.0f / (1.0f + expf(-0.5f * (Mint[i * C_ + j] + Mint[j * C_ + i])));
  }
  __syncthreads();

  float tmp[4];
#pragma unroll
  for (int k = 0; k < 4; ++k) {
    const int o = tid + k * 256;
    const int bb = o >> 6, c = o & 63;
    float a = csb[o];
    for (int j = 0; j < C_; ++j) a += csb[bb * C_ + j] * symb[j * C_ + c];
    tmp[k] = a;
  }
  __syncthreads();
#pragma unroll
  for (int k = 0; k < 4; ++k) {
    const int o = tid + k * 256;
    csb[o] = 1.0f / (1.0f + expf(-tmp[k]));
  }
  __syncthreads();

  if (tid < 64) {
    const int bb = tid >> 2, l = tid & 3;
    float a = bcls[l];
    for (int c = 0; c < C_; ++c) a += csb[bb * C_ + c] * Wcls[c * L_ + l];
    for (int h = 0; h < H_; ++h)
      a += x[(size_t)bb * S_ * H_ + h] * Wcls[(C_ + h) * L_ + l];
    out[bb * L_ + l] = a;
  }
}

// ------------------------------ launcher -----------------------------------

extern "C" void kernel_launch(void* const* d_in, const int* in_sizes, int n_in,
                              void* d_out, int out_size, void* d_ws,
                              size_t ws_size, hipStream_t stream) {
  (void)in_sizes; (void)n_in; (void)out_size; (void)ws_size;
  const float* x    = (const float*)d_in[0];
  const int*   am   = (const int*)d_in[1];
  const float* Wq   = (const float*)d_in[2];
  const float* bq   = (const float*)d_in[3];
  const float* Wk   = (const float*)d_in[4];
  const float* bk   = (const float*)d_in[5];
  const float* W1   = (const float*)d_in[6];
  const float* b1   = (const float*)d_in[7];
  const float* g1   = (const float*)d_in[8];
  const float* be1  = (const float*)d_in[9];
  const float* W2   = (const float*)d_in[10];
  const float* b2   = (const float*)d_in[11];
  const float* Wc1  = (const float*)d_in[12];
  const float* bc1  = (const float*)d_in[13];
  const float* g2   = (const float*)d_in[14];
  const float* be2  = (const float*)d_in[15];
  const float* Wc2  = (const float*)d_in[16];
  const float* bc2  = (const float*)d_in[17];
  const float* Mint = (const float*)d_in[18];
  const float* Wcls = (const float*)d_in[19];
  const float* bcls = (const float*)d_in[20];
  float* out = (float*)d_out;

  char* ws = (char*)d_ws;
  size_t off = 0;
  auto alloc = [&](size_t bytes) -> char* {
    char* p = ws + off;
    off = (off + bytes + 255) & ~(size_t)255;
    return p;
  };
  __bf16* xbf    = (__bf16*)alloc((size_t)B_ * S_ * H_ * 2);  // x row-major bf16
  __bf16* xtbf   = (__bf16*)alloc((size_t)B_ * H_ * S_ * 2);  // xT[b][h][s]
  __bf16* wqT    = (__bf16*)alloc((size_t)HH_ * H_ * 2);      // Wq^T [HH][H]
  __bf16* wkT    = (__bf16*)alloc((size_t)HH_ * H_ * 2);
  __bf16* w1T    = (__bf16*)alloc((size_t)HH_ * H_ * 2);
  __bf16* qbf    = (__bf16*)alloc((size_t)B_ * S_ * HH_ * 2);
  __bf16* kbf    = (__bf16*)alloc((size_t)B_ * S_ * HH_ * 2);
  __bf16* ctxbf  = (__bf16*)alloc((size_t)B_ * S_ * H_ * 2);
  float*  hpre   = (float*)alloc((size_t)B_ * S_ * HH_ * 4);
  float*  ts     = (float*)alloc((size_t)B_ * S_ * 4);
  float*  rmask  = (float*)alloc((size_t)B_ * S_ * 4);
  float*  pooled = (float*)alloc((size_t)B_ * H_ * 4);

  // conversions / transposes
  hipLaunchKernelGGL(cvt_bf16_kernel, dim3(4096), dim3(256), 0, stream,
                     x, xbf, (long)B_ * S_ * H_);
  hipLaunchKernelGGL(cvt_tr_kernel, dim3(H_ / 32, S_ / 32, B_), dim3(256), 0,
                     stream, x, xtbf, S_, H_);              // xT[b][h][s]
  hipLaunchKernelGGL(cvt_tr_kernel, dim3(HH_ / 32, H_ / 32, 1), dim3(256), 0,
                     stream, Wq, wqT, H_, HH_);             // WqT[c][h]
  hipLaunchKernelGGL(cvt_tr_kernel, dim3(HH_ / 32, H_ / 32, 1), dim3(256), 0,
                     stream, Wk, wkT, H_, HH_);
  hipLaunchKernelGGL(cvt_tr_kernel, dim3(HH_ / 32, H_ / 32, 1), dim3(256), 0,
                     stream, W1, w1T, H_, HH_);

  // q = x@Wq + bq, k = x@Wk + bk (bf16 outputs)
  hipLaunchKernelGGL(gemm_bf16_kernel, dim3(256, 6), dim3(256), 0, stream,
                     xbf, wqT, bq, (void*)qbf, B_ * S_, HH_, H_, 1);
  hipLaunchKernelGGL(gemm_bf16_kernel, dim3(256, 6), dim3(256), 0, stream,
                     xbf, wkT, bk, (void*)kbf, B_ * S_, HH_, H_, 1);

  // fused attention (270848 B dynamic LDS per WG; 320KB WGP LDS)
  (void)hipFuncSetAttribute(reinterpret_cast<const void*>(attn_kernel),
                            hipFuncAttributeMaxDynamicSharedMemorySize,
                            ATTN_LDS_BYTES);
  hipLaunchKernelGGL(attn_kernel, dim3(B_ * S_ / 16), dim3(256),
                     ATTN_LDS_BYTES, stream, qbf, kbf, xtbf, am, ctxbf);

  // hpre = ctx@W1 + b1 (f32)
  hipLaunchKernelGGL(gemm_bf16_kernel, dim3(256, 6), dim3(256), 0, stream,
                     ctxbf, w1T, b1, (void*)hpre, B_ * S_, HH_, H_, 0);

  hipLaunchKernelGGL(lnscore_kernel, dim3(4096), dim3(256), 0, stream,
                     hpre, g1, be1, W2, b2, am, ts);
  hipLaunchKernelGGL(span_kernel, dim3(16), dim3(256), 0, stream,
                     ts, am, rmask);
  hipLaunchKernelGGL(pool_kernel, dim3(16), dim3(256), 0, stream,
                     x, rmask, pooled);
  hipLaunchKernelGGL(head_kernel, dim3(1), dim3(256), 0, stream,
                     pooled, x, Wc1, bc1, g2, be2, Wc2, bc2, Mint, Wcls, bcls,
                     out);
}
